// Convolution_79680233275608
// MI455X (gfx1250) — compile-verified, matrix operand order checked by reading, not run on hardware
//
#include <hip/hip_runtime.h>
#include <hip/hip_bf16.h>
#include <math.h>

typedef __attribute__((ext_vector_type(16))) _Float16 v16h;
typedef __attribute__((ext_vector_type(8)))  float    v8f;

// ---------------------------------------------------------------------------
// Helpers
// ---------------------------------------------------------------------------
// Branch-free tanh-gelu using native v_exp_f32 / v_rcp_f32:
// gelu(x) = 0.5 x (1 + tanh(u)) = x * e/(e+1),  e = exp(2u),
// u = sqrt(2/pi) (x + 0.044715 x^3).
// exp2 overflow -> inf -> rcp -> 0 gives correct saturation at both ends.
__device__ __forceinline__ float gelu_f(float x) {
    const float c0 = 0.7978845608028654f;       // sqrt(2/pi)
    const float c1 = 2.8853900817779268f;       // 2*log2(e)
    float u = c0 * (x + 0.044715f * x * x * x);
    float e = __builtin_amdgcn_exp2f(u * c1);
    return x - x * __builtin_amdgcn_rcpf(e + 1.0f);
}

__device__ __forceinline__ void atomAddF(float* p, float v) {
    __hip_atomic_fetch_add(p, v, __ATOMIC_RELAXED, __HIP_MEMORY_SCOPE_AGENT);
}

__device__ __forceinline__ v8f wmma16(v16h a, v16h b, v8f c) {
    // D = A(16x32 f16) x B(32x16 f16) + C(16x16 f32)
    return __builtin_amdgcn_wmma_f32_16x16x32_f16(
        /*neg_a=*/false, a, /*neg_b=*/false, b,
        /*c_mod=*/(short)0, c, /*reuse_a=*/false, /*reuse_b=*/false);
}

// A-fragment (16x32 f16) for tile t (rows m = 16t..16t+15) of A = scale * W^T,
// where W is row-major (Kreal x 32), K zero-padded to 32.
// Layout (ISA 7.12.2): m = 16t + (lane&15), g = lane>>4,
// element e=2r+j holds K = (r>>2)*16 + 8g + 2(r&3) + j.
__device__ __forceinline__ v16h load_wT_frag(const float* __restrict__ W,
                                             int Kreal, int t, int lane,
                                             float scale) {
    int m = 16 * t + (lane & 15);
    int g = lane >> 4;
    v16h a;
#pragma unroll
    for (int e = 0; e < 16; ++e) {
        int r = e >> 1, j = e & 1;
        int k = ((r >> 2) << 4) + g * 8 + ((r & 3) << 1) + j;
        float val = (k < Kreal) ? W[k * 32 + m] * scale : 0.0f;
        a[e] = (_Float16)val;
    }
    return a;
}

// Repack two f32 C/D tiles (feature rows 0..15 / 16..31, edges on columns)
// into the f16 B-fragment of the next GEMM. Pure per-lane shuffle:
// B element e=2r+j needs K = (r>>2)*16 + 8g + 2(r&3)+j; D tile t element i
// holds feature m = 16t + 8g + i  ->  t = r>>2, i = 2(r&3)+j.
__device__ __forceinline__ v16h pack_b_from_d(v8f d0, v8f d1) {
    v16h b;
#pragma unroll
    for (int e = 0; e < 16; ++e) {
        int r = e >> 1, j = e & 1;
        int i = ((r & 3) << 1) + j;
        float val = (r >> 2) ? d1[i] : d0[i];
        b[e] = (_Float16)val;
    }
    return b;
}

// ---------------------------------------------------------------------------
// Kernel 1: node pre-transform -> feat_s (N,32), feat_v (N,32,3)
// feat = first half of node_input @ W1 / sqrt(32)
// ---------------------------------------------------------------------------
__global__ void __launch_bounds__(256)
node_pre_kernel(const float* __restrict__ node_input,
                const float* __restrict__ W1_s,
                const float* __restrict__ W1_v,
                float* __restrict__ feat_s,
                float* __restrict__ feat_v, int N) {
    __shared__ float Ws[32][32];
    __shared__ float Wv[32][32];
    for (int idx = threadIdx.x; idx < 1024; idx += blockDim.x) {
        int u = idx >> 5, w = idx & 31;
        Ws[u][w] = W1_s[u * 64 + w];   // first-half columns
        Wv[u][w] = W1_v[u * 64 + w];
    }
    __syncthreads();
    const float inv32 = 0.17677669529663687f; // 1/sqrt(32)
    int gid = blockIdx.x * blockDim.x + threadIdx.x;
    int total = N * 32;
    for (int t = gid; t < total; t += gridDim.x * blockDim.x) {
        int n = t >> 5, w = t & 31;
        const float* row = node_input + (size_t)n * 128;
        float as = 0.f, ax = 0.f, ay = 0.f, az = 0.f;
#pragma unroll 4
        for (int u = 0; u < 32; ++u) {
            as += row[u] * Ws[u][w];
            float wv = Wv[u][w];
            ax += row[32 + u * 3 + 0] * wv;
            ay += row[32 + u * 3 + 1] * wv;
            az += row[32 + u * 3 + 2] * wv;
        }
        feat_s[(size_t)n * 32 + w] = as * inv32;
        float* fv = feat_v + (size_t)n * 96 + w * 3;
        fv[0] = ax * inv32;
        fv[1] = ay * inv32;
        fv[2] = az * inv32;
    }
}

// ---------------------------------------------------------------------------
// Kernel 2: edge kernel. One wave = 16 edges. All GEMMs as W^T x X^T so the
// f32 D of one stage repacks per-lane into the f16 B of the next stage.
// Scale factors folded into the weight fragments.
// Stages: h1 = gelu((M0^T/sqrt8) es^T); h2 = gelu((M1^T/sqrt32) h1);
//         w_i = (Wtp_i^T/sqrt32) h2   (12 WMMAs / tile)
// Then elementwise tensor-product terms + atomic scatter-add to acc_s/acc_v.
// ---------------------------------------------------------------------------
__global__ void __launch_bounds__(256)
edge_kernel(const float* __restrict__ edge_attr,
            const float* __restrict__ edge_scalars,
            const float* __restrict__ M0, const float* __restrict__ M1,
            const float* __restrict__ Wtp0, const float* __restrict__ Wtp1,
            const float* __restrict__ Wtp2, const float* __restrict__ Wtp3,
            const int* __restrict__ edge_src, const int* __restrict__ edge_dst,
            const float* __restrict__ feat_s, const float* __restrict__ feat_v,
            float* __restrict__ acc_s, float* __restrict__ acc_v, int E) {
    const int lane = threadIdx.x & 31;
    const int col  = lane & 15;   // edge column within tile
    const int g    = lane >> 4;

    const float inv8  = 0.35355339059327373f;  // 1/sqrt(8)
    const float inv32 = 0.17677669529663687f;  // 1/sqrt(32)
    const float INV3  = 0.5773502691896258f;   // 1/sqrt(3)

    // Constant A-fragments (scales pre-folded; held in registers)
    v16h aM0_0 = load_wT_frag(M0, 8, 0, lane, inv8);
    v16h aM0_1 = load_wT_frag(M0, 8, 1, lane, inv8);
    v16h aM1_0 = load_wT_frag(M1, 32, 0, lane, inv32);
    v16h aM1_1 = load_wT_frag(M1, 32, 1, lane, inv32);
    v16h aT0_0 = load_wT_frag(Wtp0, 32, 0, lane, inv32);
    v16h aT0_1 = load_wT_frag(Wtp0, 32, 1, lane, inv32);
    v16h aT1_0 = load_wT_frag(Wtp1, 32, 0, lane, inv32);
    v16h aT1_1 = load_wT_frag(Wtp1, 32, 1, lane, inv32);
    v16h aT2_0 = load_wT_frag(Wtp2, 32, 0, lane, inv32);
    v16h aT2_1 = load_wT_frag(Wtp2, 32, 1, lane, inv32);
    v16h aT3_0 = load_wT_frag(Wtp3, 32, 0, lane, inv32);
    v16h aT3_1 = load_wT_frag(Wtp3, 32, 1, lane, inv32);

    int wave   = (blockIdx.x * blockDim.x + threadIdx.x) >> 5;
    int nwaves = (gridDim.x * blockDim.x) >> 5;
    int ntiles = (E + 15) >> 4;

    for (int tile = wave; tile < ntiles; tile += nwaves) {
        int e = tile * 16 + col;
        bool valid = (e < E);
        int ec = valid ? e : 0;

        // B0 from edge_scalars^T (K=8 real, padded to 32): only g==0 lanes
        // carry data (K 0..7); all other slots zero.
        v16h b0;
#pragma unroll
        for (int q = 0; q < 16; ++q) b0[q] = (_Float16)0.0f;
        if (valid && g == 0) {
            const float* p = edge_scalars + (size_t)ec * 8;
#pragma unroll
            for (int q = 0; q < 8; ++q) b0[q] = (_Float16)p[q];
        }

        v8f z = {};
        // Stage 1: (M0^T/sqrt8) x es^T, then gelu
        v8f d0 = wmma16(aM0_0, b0, z);
        v8f d1 = wmma16(aM0_1, b0, z);
#pragma unroll
        for (int q = 0; q < 8; ++q) {
            d0[q] = gelu_f(d0[q]);
            d1[q] = gelu_f(d1[q]);
        }
        v16h b1 = pack_b_from_d(d0, d1);
        // Stage 2: (M1^T/sqrt32) x h1, then gelu
        d0 = wmma16(aM1_0, b1, z);
        d1 = wmma16(aM1_1, b1, z);
#pragma unroll
        for (int q = 0; q < 8; ++q) {
            d0[q] = gelu_f(d0[q]);
            d1[q] = gelu_f(d1[q]);
        }
        v16h b2 = pack_b_from_d(d0, d1);
        // Stage 3: four tensor-product weight GEMMs (scale pre-folded)
        v8f w0a = wmma16(aT0_0, b2, z), w0b = wmma16(aT0_1, b2, z);
        v8f w1a = wmma16(aT1_0, b2, z), w1b = wmma16(aT1_1, b2, z);
        v8f w2a = wmma16(aT2_0, b2, z), w2b = wmma16(aT2_1, b2, z);
        v8f w3a = wmma16(aT3_0, b2, z), w3b = wmma16(aT3_1, b2, z);

        if (valid) {
            int src = edge_src[ec];
            int dst = edge_dst[ec];
            const float* ea = edge_attr + (size_t)ec * 4;
            float y0  = ea[0];
            float y1x = ea[1], y1y = ea[2], y1z = ea[3];
            const float* fs = feat_s + (size_t)src * 32;
            const float* fv = feat_v + (size_t)src * 96;
            float* as = acc_s + (size_t)dst * 64;
            float* av = acc_v + (size_t)dst * 192;
#pragma unroll
            for (int t2 = 0; t2 < 2; ++t2) {
#pragma unroll
                for (int i = 0; i < 8; ++i) {
                    int u = 16 * t2 + 8 * g + i;   // feature this lane owns
                    float W0 = t2 ? w0b[i] : w0a[i];
                    float W1 = t2 ? w1b[i] : w1a[i];
                    float W2 = t2 ? w2b[i] : w2a[i];
                    float W3 = t2 ? w3b[i] : w3a[i];
                    float es  = fs[u];
                    float evx = fv[u * 3 + 0];
                    float evy = fv[u * 3 + 1];
                    float evz = fv[u * 3 + 2];
                    float dotv = evx * y1x + evy * y1y + evz * y1z;
                    // out0 = [w0*es*y0 , w3*(ev.y1)/sqrt3]
                    atomAddF(as + u,      W0 * es * y0);
                    atomAddF(as + 32 + u, W3 * dotv * INV3);
                    // out1 = [w1*es*y1 , w2*ev*y0]
                    float t1 = W1 * es;
                    atomAddF(av + u * 3 + 0, t1 * y1x);
                    atomAddF(av + u * 3 + 1, t1 * y1y);
                    atomAddF(av + u * 3 + 2, t1 * y1z);
                    float t2f = W2 * y0;
                    atomAddF(av + (32 + u) * 3 + 0, t2f * evx);
                    atomAddF(av + (32 + u) * 3 + 1, t2f * evy);
                    atomAddF(av + (32 + u) * 3 + 2, t2f * evz);
                }
            }
        }
    }
}

// ---------------------------------------------------------------------------
// Kernel 3: finalize. Recompute self path (second half of W1), apply W2 to the
// aggregated node accumulators, mix, write (N,128) output.
// ---------------------------------------------------------------------------
__global__ void __launch_bounds__(256)
finalize_kernel(const float* __restrict__ node_input,
                const float* __restrict__ W1_s, const float* __restrict__ W1_v,
                const float* __restrict__ W2_s, const float* __restrict__ W2_v,
                const float* __restrict__ acc_s, const float* __restrict__ acc_v,
                float* __restrict__ out, int N) {
    __shared__ float sW2s[64][32];
    __shared__ float sW2v[64][32];
    __shared__ float sW1s[32][32];
    __shared__ float sW1v[32][32];
    for (int idx = threadIdx.x; idx < 2048; idx += blockDim.x) {
        int u = idx >> 5, w = idx & 31;
        sW2s[u][w] = W2_s[u * 32 + w];
        sW2v[u][w] = W2_v[u * 32 + w];
    }
    for (int idx = threadIdx.x; idx < 1024; idx += blockDim.x) {
        int u = idx >> 5, w = idx & 31;
        sW1s[u][w] = W1_s[u * 64 + 32 + w];   // second-half columns (self path)
        sW1v[u][w] = W1_v[u * 64 + 32 + w];
    }
    __syncthreads();
    const float inv32 = 0.17677669529663687f;          // 1/sqrt(32)
    const float cconv = 0.125f * 0.31622776601683794f; // 1/sqrt(64) * 1/sqrt(10)
    const float ca = 0.9219544457292887f;              // sqrt(1-0.15)
    const float cb = 0.3872983346207417f;              // sqrt(0.15)
    int gid = blockIdx.x * blockDim.x + threadIdx.x;
    int total = N * 32;
    for (int t = gid; t < total; t += gridDim.x * blockDim.x) {
        int n = t >> 5, w = t & 31;
        const float* row = node_input + (size_t)n * 128;
        float ss = 0.f, svx = 0.f, svy = 0.f, svz = 0.f;
#pragma unroll 4
        for (int u = 0; u < 32; ++u) {
            ss += row[u] * sW1s[u][w];
            float wv = sW1v[u][w];
            svx += row[32 + u * 3 + 0] * wv;
            svy += row[32 + u * 3 + 1] * wv;
            svz += row[32 + u * 3 + 2] * wv;
        }
        ss *= inv32; svx *= inv32; svy *= inv32; svz *= inv32;
        const float* as = acc_s + (size_t)n * 64;
        const float* av = acc_v + (size_t)n * 192;
        float cs = 0.f, cvx = 0.f, cvy = 0.f, cvz = 0.f;
#pragma unroll 4
        for (int u = 0; u < 64; ++u) {
            cs += as[u] * sW2s[u][w];
            float wv = sW2v[u][w];
            cvx += av[u * 3 + 0] * wv;
            cvy += av[u * 3 + 1] * wv;
            cvz += av[u * 3 + 2] * wv;
        }
        cs *= cconv; cvx *= cconv; cvy *= cconv; cvz *= cconv;
        float* o = out + (size_t)n * 128;
        o[w] = ca * ss + cb * cs;
        o[32 + w * 3 + 0] = ca * svx + cb * cvx;
        o[32 + w * 3 + 1] = ca * svy + cb * cvy;
        o[32 + w * 3 + 2] = ca * svz + cb * cvz;
    }
}

// ---------------------------------------------------------------------------
// Launcher
// ---------------------------------------------------------------------------
extern "C" void kernel_launch(void* const* d_in, const int* in_sizes, int n_in,
                              void* d_out, int out_size, void* d_ws, size_t ws_size,
                              hipStream_t stream) {
    (void)n_in; (void)out_size; (void)ws_size;
    const float* node_input   = (const float*)d_in[0];
    const float* edge_attr    = (const float*)d_in[1];
    const float* edge_scalars = (const float*)d_in[2];
    const float* W1_s = (const float*)d_in[3];
    const float* W1_v = (const float*)d_in[4];
    const float* M0   = (const float*)d_in[5];
    const float* M1   = (const float*)d_in[6];
    const float* Wtp0 = (const float*)d_in[7];
    const float* Wtp1 = (const float*)d_in[8];
    const float* Wtp2 = (const float*)d_in[9];
    const float* Wtp3 = (const float*)d_in[10];
    const float* W2_s = (const float*)d_in[11];
    const float* W2_v = (const float*)d_in[12];
    const int* edge_src = (const int*)d_in[13];
    const int* edge_dst = (const int*)d_in[14];

    int N = in_sizes[0] / 128;
    int E = in_sizes[13];

    // Workspace layout (floats): feat_s[N*32] | feat_v[N*96] | acc_s[N*64] | acc_v[N*192]
    float* feat_s = (float*)d_ws;
    float* feat_v = feat_s + (size_t)N * 32;
    float* acc_s  = feat_v + (size_t)N * 96;
    float* acc_v  = acc_s  + (size_t)N * 64;

    hipMemsetAsync(acc_s, 0, (size_t)N * 256 * sizeof(float), stream);

    const int threads = 256;
    int blocksN = (N * 32 + threads - 1) / threads;
    node_pre_kernel<<<blocksN, threads, 0, stream>>>(node_input, W1_s, W1_v,
                                                     feat_s, feat_v, N);

    int blocksE = 1024;  // 8192 waves grid-striding over E/16 tiles
    edge_kernel<<<blocksE, threads, 0, stream>>>(edge_attr, edge_scalars,
                                                 M0, M1, Wtp0, Wtp1, Wtp2, Wtp3,
                                                 edge_src, edge_dst,
                                                 feat_s, feat_v, acc_s, acc_v, E);

    finalize_kernel<<<blocksN, threads, 0, stream>>>(node_input, W1_s, W1_v,
                                                     W2_s, W2_v, acc_s, acc_v,
                                                     (float*)d_out, N);
}
